// BilstmCRF_81612968559234
// MI455X (gfx1250) — compile-verified, matrix operand order checked by reading, not run on hardware
//
#include <hip/hip_runtime.h>
#include <hip/hip_bf16.h>
#include <stdint.h>
#include <stddef.h>

typedef __attribute__((ext_vector_type(16))) __bf16 v16bf;
typedef __attribute__((ext_vector_type(8)))  float  v8f;
typedef __attribute__((ext_vector_type(4)))  unsigned int v4u;
typedef __attribute__((ext_vector_type(8)))  int v8i;
typedef __attribute__((ext_vector_type(4)))  int v4i;

namespace {
constexpr int E_ = 256, H_ = 512, NTAG = 16, B_ = 64, T_ = 1024;
constexpr int BT_ = B_ * T_;
constexpr int G4H = 4 * H_;          // 2048
constexpr int BOSt = 1, EOSt = 2;
constexpr int NBLK = 64;             // persistent LSTM blocks (2 dirs x 32 j-tiles)
}

// ---- ISA 16-bit A-fragment K mapping (CDNA5 ISA, 16x32 bf16) ----
__device__ __forceinline__ int frag_k(int e, int half) {
  return e < 8 ? half * 8 + e : 16 + half * 8 + (e - 8);
}
__device__ __forceinline__ void frag_pos(int kk, int& half, int& e) {
  if (kk < 16) { half = kk >> 3; e = kk & 7; }
  else         { int k2 = kk - 16; half = k2 >> 3; e = 8 + (k2 & 7); }
}

// ---------------------------------------------------------------- zero init
__global__ void zero_kernel(float4* __restrict__ p, int n4) {
  int i = blockIdx.x * blockDim.x + threadIdx.x;
  if (i < n4) p[i] = make_float4(0.f, 0.f, 0.f, 0.f);
}

// ------------------------------------------------- weight pack: W(N,K) -> Bfrag
__global__ void pack_B_frag(const float* __restrict__ W, __bf16* __restrict__ dst,
                            int N, int K) {
  int tid = blockIdx.x * blockDim.x + threadIdx.x;
  int total = (K >> 5) * (N >> 4) * 512;
  if (tid >= total) return;
  int e = tid & 15;
  int lane = (tid >> 4) & 31;
  int frag = tid >> 9;
  int ntiles = N >> 4;
  int n_tile = frag % ntiles;
  int k_tile = frag / ntiles;
  int n = (n_tile << 4) + (lane & 15);
  int k = (k_tile << 5) + frag_k(e, lane >> 4);
  dst[tid] = (__bf16)W[(size_t)n * K + k];
}

// --------------------------------- embedding gather fused with A-fragment pack
__global__ void embed_pack(const int* __restrict__ x, const float* __restrict__ emb,
                           __bf16* __restrict__ Xf) {
  int tid = blockIdx.x * blockDim.x + threadIdx.x;   // exact: BT/16 * E/32 * 512
  int e = tid & 15;
  int lane = (tid >> 4) & 31;
  int frag = tid >> 9;
  constexpr int KT = E_ >> 5;                        // 8
  int k_tile = frag % KT;
  int m_tile = frag / KT;
  int m = (m_tile << 4) + (lane & 15);
  int k = (k_tile << 5) + frag_k(e, lane >> 4);
  int tok = x[m];
  Xf[tid] = (__bf16)emb[(size_t)tok * E_ + k];
}

// --------------- input projection: xp = X @ W_ih^T + b ; 4 N-tiles per wave
__global__ __launch_bounds__(32) void proj_gemm(
    const __bf16* __restrict__ Xf,
    const __bf16* __restrict__ BfF, const __bf16* __restrict__ BfB,
    const float* __restrict__ biasF, const float* __restrict__ biasB,
    __bf16* __restrict__ xpF, __bf16* __restrict__ xpB) {
  const int lane = threadIdx.x;
  const int nq = blockIdx.x;         // 0..31 -> n_tiles 4*nq .. 4*nq+3
  const int m_tile = blockIdx.y;     // 0..4095
  const int dir = blockIdx.z;
  const __bf16* Bf = dir ? BfB : BfF;
  const float* bias = dir ? biasB : biasF;
  __bf16* out = dir ? xpB : xpF;
  constexpr int KT = E_ >> 5;        // 8
  constexpr int NTIL = G4H >> 4;     // 128
  v8f acc[4] = {};
#pragma unroll
  for (int kt = 0; kt < KT; ++kt) {
    v16bf a = *(const v16bf*)(Xf + (((size_t)m_tile * KT + kt) * 32 + lane) * 16);
    if (kt + 1 < KT)
      __builtin_prefetch(Xf + (((size_t)m_tile * KT + kt + 1) * 32 + lane) * 16, 0, 3);
#pragma unroll
    for (int i = 0; i < 4; ++i) {
      v16bf b = *(const v16bf*)(Bf + (((size_t)kt * NTIL + nq * 4 + i) * 32 + lane) * 16);
      acc[i] = __builtin_amdgcn_wmma_f32_16x16x32_bf16(false, a, false, b, (short)0,
                                                       acc[i], false, false);
    }
  }
  const int half = lane >> 4, col = lane & 15;
#pragma unroll
  for (int i = 0; i < 4; ++i) {
    const int n = ((nq * 4 + i) << 4) + col;
    const float bv = bias[n];
#pragma unroll
    for (int r = 0; r < 8; ++r) {
      int m = (m_tile << 4) + r + (half << 3);
      out[(size_t)m * G4H + n] = (__bf16)(acc[i][r] + bv);
    }
  }
}

// ----------------------------------------------------------- grid barrier
__device__ __forceinline__ void grid_barrier(int* ctr, int target) {
  __syncthreads();
  if (threadIdx.x == 0) {
    __threadfence();   // release all prior stores to device scope
    __hip_atomic_fetch_add(ctr, 1, __ATOMIC_RELEASE, __HIP_MEMORY_SCOPE_AGENT);
    while (__hip_atomic_load(ctr, __ATOMIC_ACQUIRE, __HIP_MEMORY_SCOPE_AGENT) < target)
      __builtin_amdgcn_s_sleep(1);
  }
  __syncthreads();
  __threadfence();     // acquire: see other blocks' stores
}

// ------------- persistent bidirectional LSTM: all T steps in one kernel.
// grid = 64 blocks (dir[2] x j[32]), block = 128 threads (4 waves = m_tiles).
// w_hh B-fragments (64 KB per block) staged once into LDS via the Tensor Data
// Mover; (h,c) cell state lives in registers; h fragments ping-pong through L2.
__global__ __launch_bounds__(128) void lstm_persistent(
    const int* __restrict__ x,
    const __bf16* __restrict__ xpF, const __bf16* __restrict__ xpB,
    const __bf16* __restrict__ BhhF, const __bf16* __restrict__ BhhB,
    __bf16* __restrict__ hf0, __bf16* __restrict__ hf1,   // [dir][4][16][512]
    __bf16* __restrict__ hEmis, int* __restrict__ ctr) {
  __shared__ __bf16 Bsh[64 * 512];          // 64 fragments (4 gates x 16 kt) = 64 KB
  const int lane = threadIdx.x & 31;
  const int m_tile = threadIdx.x >> 5;       // 0..3
  const int dir = blockIdx.x >> 5;
  const int j = blockIdx.x & 31;             // hidden tile
  const __bf16* xp = dir ? xpB : xpF;
  const __bf16* Bhh = dir ? BhhB : BhhF;
  const size_t doff = (size_t)dir * 32768;   // dir offset into h fragment buffers
  constexpr int KT = H_ >> 5;                // 16

  // ---- TDM: stage this block's w_hh fragments into LDS (3D tile:
  // x = 512 elem fragment, y = 16 k-tiles (stride 65536 elem), z = 4 gates
  // (stride 16384 elem)). LDS iteration order == (gate, kt, frag) layout.
  if (m_tile == 0) {
    unsigned long long gaddr =
        (unsigned long long)(const char*)Bhh + (unsigned long long)j * 1024ull;
    unsigned int ldsaddr = (unsigned int)(size_t)(&Bsh[0]);
    v4u g0;
    g0[0] = 1u;                                           // count=1
    g0[1] = ldsaddr;                                      // lds_addr
    g0[2] = (unsigned int)(gaddr & 0xFFFFFFFFull);        // global_addr[31:0]
    g0[3] = (unsigned int)((gaddr >> 32) & 0x01FFFFFFull) // global_addr[56:32]
            | (2u << 30);                                 // type=2 (image)
    v8i g1;
    g1[0] = 0x00010000;                // data_size=1 (2 bytes), no mask/flags
    g1[1] = (int)(512u << 16);         // tensor_dim0[15:0] = 512
    g1[2] = (int)(16u << 16);          // dim0 hi=0 | tensor_dim1[15:0]=16
    g1[3] = (int)(512u << 16);         // dim1 hi=0 | tile_dim0=512
    g1[4] = 16 | (4 << 16);            // tile_dim1=16 | tile_dim2=4
    g1[5] = 65536;                     // tensor_dim0_stride lo
    g1[6] = (int)(16384u << 16);       // stride0 hi=0 | tensor_dim1_stride lo16
    g1[7] = 0;                         // tensor_dim1_stride hi
    v4i g2;
    g2[0] = 4;                         // tensor_dim2 = 4
    g2[1] = 1;                         // tensor_dim3 (unused, tile_dim3=0)
    g2[2] = 0;
    g2[3] = 0;                         // tile_dim3 = 0
    v4i g3 = {0, 0, 0, 0};
    v8i g4 = {0, 0, 0, 0, 0, 0, 0, 0}; // extra group on clang-23 6-arg form
    __builtin_amdgcn_tensor_load_to_lds(g0, g1, g2, g3, g4, 0);
    __builtin_amdgcn_s_wait_tensorcnt(0);
  }
  __syncthreads();

  const int half = lane >> 4, col = lane & 15;
  const int u = (j << 4) + col;              // hidden unit
  float hreg[8], creg[8];
#pragma unroll
  for (int r = 0; r < 8; ++r) { hreg[r] = 0.f; creg[r] = 0.f; }

  for (int s = 0; s < T_; ++s) {
    const __bf16* hfrd = ((s & 1) ? hf1 : hf0) + doff;
    __bf16* hfwr = ((s & 1) ? hf0 : hf1) + doff;
    const int t = dir ? (T_ - 1 - s) : s;

    v8f ai = {}, af = {}, ag = {}, ao = {};
    for (int kt = 0; kt < KT; ++kt) {
      v16bf a  = *(const v16bf*)(hfrd + (((size_t)m_tile * KT + kt) * 32 + lane) * 16);
      v16bf b0 = *(const v16bf*)(Bsh + (((0 * 16 + kt) * 32 + lane) * 16));
      v16bf b1 = *(const v16bf*)(Bsh + (((1 * 16 + kt) * 32 + lane) * 16));
      v16bf b2 = *(const v16bf*)(Bsh + (((2 * 16 + kt) * 32 + lane) * 16));
      v16bf b3 = *(const v16bf*)(Bsh + (((3 * 16 + kt) * 32 + lane) * 16));
      ai = __builtin_amdgcn_wmma_f32_16x16x32_bf16(false, a, false, b0, (short)0, ai, false, false);
      af = __builtin_amdgcn_wmma_f32_16x16x32_bf16(false, a, false, b1, (short)0, af, false, false);
      ag = __builtin_amdgcn_wmma_f32_16x16x32_bf16(false, a, false, b2, (short)0, ag, false, false);
      ao = __builtin_amdgcn_wmma_f32_16x16x32_bf16(false, a, false, b3, (short)0, ao, false, false);
    }

#pragma unroll
    for (int r = 0; r < 8; ++r) {
      const int m = (m_tile << 4) + r + (half << 3);     // batch index
      const size_t row = (size_t)m * T_ + t;             // (b,t) flat
      const size_t xb = row * (size_t)G4H;
      float zi = ai[r] + (float)xp[xb + u];
      float zf = af[r] + (float)xp[xb + H_ + u];
      float zg = ag[r] + (float)xp[xb + 2 * H_ + u];
      float zo = ao[r] + (float)xp[xb + 3 * H_ + u];
      float gi = 1.f / (1.f + __expf(-zi));
      float gf = 1.f / (1.f + __expf(-zf));
      float go = 1.f / (1.f + __expf(-zo));
      float gg = tanhf(zg);
      float cp = creg[r], hp = hreg[r];
      float cn = gf * cp + gi * gg;
      float hn = go * tanhf(cn);
      float mt = (x[row] != 0) ? 1.f : 0.f;
      float h2 = mt * hn + (1.f - mt) * hp;
      float c2 = mt * cn + (1.f - mt) * cp;
      hreg[r] = h2;
      creg[r] = c2;
      {   // next-step A fragment (M=batch, K=hidden)
        int kk = u & 31, kt2 = u >> 5, hh, ee;
        frag_pos(kk, hh, ee);
        int alane = (m & 15) + (hh << 4);
        hfwr[(((size_t)(m >> 4) * KT + kt2) * 32 + alane) * 16 + ee] = (__bf16)h2;
      }
      {   // emissions A fragment (M=b*T+t, K=dir*H+u), value = mask*h_n
        int kg = dir * H_ + u;
        int kk = kg & 31, kt2 = kg >> 5, hh, ee;
        frag_pos(kk, hh, ee);
        int elane = ((int)(row & 15)) + (hh << 4);
        hEmis[(((size_t)(row >> 4) * (2 * H_ >> 5) + kt2) * 32 + elane) * 16 + ee] =
            (__bf16)(mt * hn);
      }
    }
    grid_barrier(ctr, NBLK * (s + 1));
  }
}

// ---------- emissions = hiddens @ w_fc^T + b_fc ; 4 M-tiles per wave (N=16)
__global__ __launch_bounds__(32) void emis_gemm(
    const __bf16* __restrict__ Af, const __bf16* __restrict__ Bfc,
    const float* __restrict__ bfc, float* __restrict__ emis) {
  const int lane = threadIdx.x;
  const int mq = blockIdx.x;          // 0..1023 -> m_tiles 4*mq .. 4*mq+3
  constexpr int KT = (2 * H_) >> 5;   // 32
  v8f acc[4] = {};
  for (int kt = 0; kt < KT; ++kt) {
    v16bf b = *(const v16bf*)(Bfc + ((size_t)kt * 32 + lane) * 16);
#pragma unroll
    for (int i = 0; i < 4; ++i) {
      v16bf a =
          *(const v16bf*)(Af + (((size_t)(4 * mq + i) * KT + kt) * 32 + lane) * 16);
      acc[i] = __builtin_amdgcn_wmma_f32_16x16x32_bf16(false, a, false, b, (short)0,
                                                       acc[i], false, false);
    }
  }
  const int half = lane >> 4, col = lane & 15;
  const float bv = bfc[col];
#pragma unroll
  for (int i = 0; i < 4; ++i)
#pragma unroll
    for (int r = 0; r < 8; ++r) {
      int m = ((4 * mq + i) << 4) + r + (half << 3);
      emis[(size_t)m * NTAG + col] = acc[i][r] + bv;
    }
}

// ------------------------------------------------ Viterbi: 1 wave per batch row
__global__ void viterbi_kernel(const int* __restrict__ x, const float* __restrict__ emis,
                               const float* __restrict__ trans, int* __restrict__ bp,
                               float* __restrict__ out) {
  const int b = blockIdx.x;
  const int lane = threadIdx.x;
  __shared__ float tr[NTAG][NTAG];
  if (lane < NTAG)
    for (int i = 0; i < NTAG; ++i) tr[i][lane] = trans[i * NTAG + lane];
  __syncthreads();

  const int n = lane & 15;
  float alpha = tr[BOSt][n] + emis[((size_t)b * T_ + 0) * NTAG + n];
  for (int t = 1; t < T_; ++t) {
    float e = emis[((size_t)b * T_ + t) * NTAG + n];
    float best = -3.4e38f;
    int barg = 0;
    for (int i = 0; i < NTAG; ++i) {
      float aprev = __shfl(alpha, i, 32);
      float s = aprev + tr[i][n];
      if (s > best) { best = s; barg = i; }
    }
    best += e;
    int mt = (x[(size_t)b * T_ + t] != 0);
    float an = mt ? best : alpha;
    int bpn = mt ? barg : n;
    alpha = an;
    if (lane < NTAG) bp[((size_t)b * (T_ - 1) + (t - 1)) * NTAG + n] = bpn;
  }
  float endv = alpha + tr[n][EOSt];
  float bestv = endv;
  int bestt = n;
  for (int off = 8; off > 0; off >>= 1) {
    float ov = __shfl_down(bestv, off, 16);
    int ot = __shfl_down(bestt, off, 16);
    if (ov > bestv || (ov == bestv && ot < bestt)) { bestv = ov; bestt = ot; }
  }
  if (lane == 0) {
    out[b] = bestv;                     // score
    float* path = out + B_;             // path (as float), B*T
    int tag = bestt;
    int mT = (x[(size_t)b * T_ + (T_ - 1)] != 0);
    path[(size_t)b * T_ + (T_ - 1)] = (float)(mT ? tag : 0);
    for (int t = T_ - 2; t >= 0; --t) {
      tag = bp[((size_t)b * (T_ - 1) + t) * NTAG + tag];
      int mm = (x[(size_t)b * T_ + t] != 0);
      path[(size_t)b * T_ + t] = (float)(mm ? tag : 0);
    }
  }
}

// =============================================================== host driver
extern "C" void kernel_launch(void* const* d_in, const int* in_sizes, int n_in,
                              void* d_out, int out_size, void* d_ws, size_t ws_size,
                              hipStream_t stream) {
  const int*   x      = (const int*)d_in[0];
  const float* emb    = (const float*)d_in[1];
  const float* w_ih_f = (const float*)d_in[2];
  const float* w_hh_f = (const float*)d_in[3];
  const float* b_f    = (const float*)d_in[4];
  const float* w_ih_b = (const float*)d_in[5];
  const float* w_hh_b = (const float*)d_in[6];
  const float* b_b    = (const float*)d_in[7];
  const float* w_fc   = (const float*)d_in[8];
  const float* b_fc   = (const float*)d_in[9];
  const float* trans  = (const float*)d_in[10];

  char* p = (char*)d_ws;
  auto take = [&](size_t bytes) { char* q = p; p += bytes; return q; };
  constexpr size_t SZ_BIH   = (size_t)(E_ / 32) * (G4H / 16) * 512 * 2;       // 1 MB
  constexpr size_t SZ_BHH   = (size_t)(H_ / 32) * (G4H / 16) * 512 * 2;       // 2 MB
  constexpr size_t SZ_BFC   = (size_t)(2 * H_ / 32) * (NTAG / 16) * 512 * 2;  // 32 KB
  constexpr size_t SZ_XFRAG = (size_t)(BT_ / 16) * (E_ / 32) * 512 * 2;       // 32 MB
  constexpr size_t SZ_XP    = (size_t)BT_ * G4H * 2;                          // 256 MB
  constexpr size_t SZ_HFRAG = 2 * (size_t)(B_ / 16) * (H_ / 32) * 512 * 2;    // 128 KB (2 dirs)
  constexpr size_t SZ_HEMIS = (size_t)(BT_ / 16) * (2 * H_ / 32) * 512 * 2;   // 128 MB
  constexpr size_t SZ_EMIS  = (size_t)BT_ * NTAG * 4;                         // 4 MB
  constexpr size_t SZ_BP    = (size_t)B_ * (T_ - 1) * NTAG * 4;               // 4 MB

  __bf16* BIHF = (__bf16*)take(SZ_BIH);
  __bf16* BIHB = (__bf16*)take(SZ_BIH);
  __bf16* BHHF = (__bf16*)take(SZ_BHH);
  __bf16* BHHB = (__bf16*)take(SZ_BHH);
  __bf16* BFC  = (__bf16*)take(SZ_BFC);
  __bf16* XFR  = (__bf16*)take(SZ_XFRAG);
  __bf16* XPF  = (__bf16*)take(SZ_XP);
  __bf16* XPB  = (__bf16*)take(SZ_XP);
  char* zero_base = p;                        // contiguous zero region
  __bf16* HF0 = (__bf16*)take(SZ_HFRAG);
  __bf16* HF1 = (__bf16*)take(SZ_HFRAG);
  int* CTR = (int*)take(16);
  size_t zero_bytes = (size_t)(p - zero_base);
  __bf16* HEMIS = (__bf16*)take(SZ_HEMIS);
  float* EMIS = (float*)take(SZ_EMIS);
  int* BP = (int*)take(SZ_BP);

  {  // zero h fragments + barrier counter (re-done every call -> replay-safe)
    int n4 = (int)(zero_bytes / 16);
    zero_kernel<<<(n4 + 255) / 256, 256, 0, stream>>>((float4*)zero_base, n4);
  }

  auto packN = [&](const float* W, __bf16* dst, int N, int K) {
    int total = (K / 32) * (N / 16) * 512;
    pack_B_frag<<<(total + 255) / 256, 256, 0, stream>>>(W, dst, N, K);
  };
  packN(w_ih_f, BIHF, G4H, E_);
  packN(w_ih_b, BIHB, G4H, E_);
  packN(w_hh_f, BHHF, G4H, H_);
  packN(w_hh_b, BHHB, G4H, H_);
  packN(w_fc, BFC, NTAG, 2 * H_);

  {
    int total = (BT_ / 16) * (E_ / 32) * 512;
    embed_pack<<<total / 256, 256, 0, stream>>>(x, emb, XFR);
  }

  proj_gemm<<<dim3(32, BT_ / 16, 2), 32, 0, stream>>>(XFR, BIHF, BIHB, b_f, b_b, XPF,
                                                      XPB);

  lstm_persistent<<<NBLK, 128, 0, stream>>>(x, XPF, XPB, BHHF, BHHB, HF0, HF1, HEMIS,
                                            CTR);

  emis_gemm<<<BT_ / 16 / 4, 32, 0, stream>>>(HEMIS, BFC, b_fc, EMIS);

  viterbi_kernel<<<B_, 32, 0, stream>>>(x, EMIS, trans, BP, (float*)d_out);
}
